// MDnet_14044543058312
// MI455X (gfx1250) — matrix operand order repaired and checked by reading, not compiled, not run on hardware
//
#include <hip/hip_runtime.h>
#include <hip/hip_bf16.h>

// ---------------------------------------------------------------------------
// MDnet forward for MI455X (gfx1250, wave32, WMMA).
// All heavy GEMMs use v_wmma_f32_16x16x32_f16 (f16 in, f32 acc).
// Fragment loads are vectorized to 128-bit (global_load_b128 / ds_read_b128).
// Workspace use ~116 MB of d_ws. No atomics (deterministic).
// ---------------------------------------------------------------------------

typedef _Float16 half_t;
typedef _Float16 v16h __attribute__((ext_vector_type(16)));
typedef _Float16 v8h  __attribute__((ext_vector_type(8)));
typedef float    v8f  __attribute__((ext_vector_type(8)));

constexpr int B_   = 8;
constexpr int TT   = 4000;
constexpr int L_   = 2048;
constexpr int D_   = 128;
constexpr int TS   = 2064;   // padded time-stride for conv buffers (mult of 16)
constexpr int TE   = 2051;   // emb columns needed by the conv chain

#define WMMA_F16(a,b,c) \
  __builtin_amdgcn_wmma_f32_16x16x32_f16(false,(a),false,(b),(short)0,(c),false,false)

// ---- WMMA fragment loaders (contiguous k order, same for A and B), 128-bit vector loads -----
// A (16x32 f16): lane m = l&15; halfs j -> k = kk + hi*16 + j   (hi = lane>=16)
__device__ __forceinline__ v16h frag_a_f16(const half_t* row, int kk, int hi) {
  const half_t* p0 = row + kk + (hi ? 16 : 0);
  v8h lo = *(const v8h*)(p0);
  v8h up = *(const v8h*)(p0 + 8);
  return __builtin_shufflevector(lo, up, 0,1,2,3,4,5,6,7,8,9,10,11,12,13,14,15);
}
__device__ __forceinline__ v16h frag_a_f32(const float* row, int kk, int hi) {
  const float* p0 = row + kk + (hi ? 16 : 0);
  v8h lo = __builtin_convertvector(*(const v8f*)(p0),      v8h);
  v8h up = __builtin_convertvector(*(const v8f*)(p0 + 8), v8h);
  return __builtin_shufflevector(lo, up, 0,1,2,3,4,5,6,7,8,9,10,11,12,13,14,15);
}
// B (32x16 f16) supplied as W^T where W is row-major [N][K]:
// lane n = l&15; halfs j -> k = kk + hi*16 + j (32 contiguous bytes).
__device__ __forceinline__ v16h frag_b_row(const half_t* wrow, int kk, int hi) {
  const half_t* p = wrow + kk + (hi ? 16 : 0);
  v8h lo = *(const v8h*)(p);
  v8h up = *(const v8h*)(p + 8);
  return __builtin_shufflevector(lo, up, 0,1,2,3,4,5,6,7,8,9,10,11,12,13,14,15);
}

__device__ __forceinline__ float sigmoidf_(float x) { return 1.f / (1.f + __expf(-x)); }

// ---- weight prep ----------------------------------------------------------
__global__ void castf_k(const float* __restrict__ s, half_t* __restrict__ d, int n) {
  int i = blockIdx.x * blockDim.x + threadIdx.x;
  if (i < n) d[i] = (half_t)s[i];
}
__global__ void wae_k(const float* __restrict__ Wa_w, half_t* __restrict__ d) {
  int i = blockIdx.x * blockDim.x + threadIdx.x;
  if (i < 256 * 256) { int n = i >> 8, k = i & 255; d[i] = (half_t)Wa_w[n * 512 + 256 + k]; }
}
__global__ void convw_k(const float* __restrict__ cw, half_t* __restrict__ d) {
  int i = blockIdx.x * blockDim.x + threadIdx.x;
  if (i < 3 * 128 * 256) {
    int layer = i / 32768, r = i % 32768, co = r >> 8, k = r & 255;
    int ci = k & 127, tap = k >> 7;
    d[i] = (half_t)cw[(((size_t)layer * 128 + co) * 128 + ci) * 2 + tap];
  }
}

// ---- embedding lookup: embT (channel-major, unmasked) + padh (masked) -----
__global__ void embed_k(const int* __restrict__ tok, const int* __restrict__ lens,
                        const float* __restrict__ embed,
                        half_t* __restrict__ embT, half_t* __restrict__ padh) {
  int t = blockIdx.x, b = blockIdx.y, c = threadIdx.x;  // c in [0,128)
  int id = tok[b * TT + t];
  float v = embed[(size_t)id * D_ + c];
  embT[(size_t)(b * D_ + c) * TS + t] = (half_t)v;
  if (t < L_) {
    int len = lens[b];
    padh[((size_t)(b * L_ + t)) * D_ + c] = (half_t)((t < len) ? v : 0.f);
  }
}

// ---- conv layer as GEMM: out[b,co,t] = relu(bias + W[co,:]·x2[:,t]) -------
// block = 256 (8 waves), wave w handles cout tile w. grid = (ceil(Lout/16), B)
__global__ void conv_wmma_k(const half_t* __restrict__ X, const half_t* __restrict__ Wc,
                            const float* __restrict__ bias, half_t* __restrict__ Y, int Lout) {
  const int lane = threadIdx.x & 31, wave = threadIdx.x >> 5;
  const int hi = lane >> 4;
  const int t0 = blockIdx.x * 16, b = blockIdx.y;
  const half_t* arow = Wc + (size_t)(wave * 16 + (lane & 15)) * 256;
  const int n = lane & 15;
  int t = t0 + n;
  int tc = (t < Lout) ? t : 0;  // clamp (discarded columns)
  const half_t* xb = X + (size_t)b * 128 * TS;
  v8f acc = {};
  for (int kk = 0; kk < 256; kk += 32) {
    v16h a = frag_a_f16(arow, kk, hi);
    v16h bb;
    const int kb = kk + hi * 16;
#pragma unroll
    for (int j = 0; j < 16; ++j) {
      int k = kb + j;
      bb[j] = xb[(size_t)(k & 127) * TS + tc + (k >> 7)];
    }
    acc = WMMA_F16(a, bb, acc);
  }
  if (t < Lout) {
#pragma unroll
    for (int j = 0; j < 8; ++j) {
      int co = wave * 16 + j + hi * 8;
      float v = acc[j] + bias[co];
      Y[(size_t)(b * 128 + co) * TS + t] = (half_t)(v > 0.f ? v : 0.f);
    }
  }
}

// ---- masked mean over time of conv3 output --------------------------------
__global__ void tmean_k(const half_t* __restrict__ C3, const int* __restrict__ lens,
                        float* __restrict__ tmean) {
  int c = blockIdx.x, b = blockIdx.y, lane = threadIdx.x;
  const half_t* p = C3 + (size_t)(b * 128 + c) * TS;
  int len = lens[b];
  float s = 0.f;
  for (int t = lane; t < L_; t += 32) s += (t < len) ? (float)p[t] : 0.f;
  for (int o = 16; o > 0; o >>= 1) s += __shfl_xor(s, o, 32);
  if (lane == 0) tmean[b * 128 + c] = s * (1.f / 2048.f);
}

// ---- generic WMMA GEMM: Out[m, n] = oscale * (A[m,:] · W[n,:]) + bias[n] --
// AMODE 0: A f16 [B,L,128]    (K=128)
// AMODE 1: A f16 [B,L,128], duplicated features (K=256, reads k&127)
// AMODE 2: A f16 [B,L,256]    (K=256)
// AMODE 3: A f32 [B,L,256]    (K=256, convert on load)
// REV: gather source row via packed-reverse index (backward GRU direction)
template <int AMODE, bool REV, bool OUTF16>
__global__ void gemm_wmma(const void* __restrict__ Asrc, const half_t* __restrict__ W,
                          int ldw, const float* __restrict__ bias, float oscale,
                          void* __restrict__ Out, int ldo, int K,
                          const int* __restrict__ lens) {
  const int lane = threadIdx.x;
  const int hi = lane >> 4;
  const int mt = blockIdx.x, nt = blockIdx.y;
  int m = mt * 16 + (lane & 15);
  int b = m >> 11, l = m & 2047;
  if (REV) { int len = lens[b]; l = (l < len) ? (len - 1 - l) : l; }
  const half_t* arow16 = nullptr;
  const float*  arow32 = nullptr;
  if (AMODE == 0 || AMODE == 1) arow16 = (const half_t*)Asrc + ((size_t)(b * L_ + l)) * 128;
  else if (AMODE == 2)          arow16 = (const half_t*)Asrc + ((size_t)(b * L_ + l)) * 256;
  else                          arow32 = (const float*)Asrc + ((size_t)(b * L_ + l)) * 256;
  const int n = nt * 16 + (lane & 15);
  const half_t* wrow = W + (size_t)n * ldw;
  v8f acc = {};
  for (int kk = 0; kk < K; kk += 32) {
    int ka = (AMODE == 1) ? (kk & 127) : kk;
    v16h a;
    if (AMODE == 3) a = frag_a_f32(arow32, ka, hi);
    else            a = frag_a_f16(arow16, ka, hi);
    v16h bb = frag_b_row(wrow, kk, hi);
    acc = WMMA_F16(a, bb, acc);
  }
  float bs = bias ? bias[n] : 0.f;
#pragma unroll
  for (int j = 0; j < 8; ++j) {
    int row = mt * 16 + j + hi * 8;
    size_t idx = (size_t)row * ldo + n;
    float v = acc[j] * oscale + bs;
    if (OUTF16) ((half_t*)Out)[idx] = (half_t)v;
    else        ((float*)Out)[idx] = v;
  }
}

// ---- persistent GRU scan: one block per direction, h in LDS ---------------
template <bool OUTF32>
__global__ void gru_scan(const half_t* __restrict__ xgf, const half_t* __restrict__ xgb,
                         const half_t* __restrict__ Whh, const float* __restrict__ bhh,
                         void* __restrict__ out, const int* __restrict__ lens) {
  const int dir = blockIdx.x;
  const half_t* xg = dir ? xgb : xgf;
  const half_t* Wd = Whh + (size_t)dir * 384 * 128;
  const float*  bd = bhh + dir * 384;
  const int colbase = dir * 128;
  __shared__ half_t h16[16 * 128];
  __shared__ float  hf[8 * 128];
  __shared__ float  hhacc[16 * 384];
  for (int i = threadIdx.x; i < 16 * 128; i += blockDim.x) h16[i] = (half_t)0.f;
  for (int i = threadIdx.x; i < 8 * 128; i += blockDim.x) hf[i] = 0.f;
  __syncthreads();
  const int lane = threadIdx.x & 31, wave = threadIdx.x >> 5;
  const int hi = lane >> 4;
  for (int l = 0; l < L_; ++l) {
    // hh = h16 @ Whh^T  -- 24 N-tiles over 8 waves, K=128
#pragma unroll
    for (int u = 0; u < 3; ++u) {
      int n = (wave * 3 + u) * 16 + (lane & 15);
      const half_t* wrow = Wd + (size_t)n * 128;
      const half_t* arow = h16 + (size_t)(lane & 15) * 128;
      v8f acc = {};
#pragma unroll
      for (int kk = 0; kk < 128; kk += 32)
        acc = WMMA_F16(frag_a_f16(arow, kk, hi), frag_b_row(wrow, kk, hi), acc);
#pragma unroll
      for (int j = 0; j < 8; ++j) hhacc[(j + hi * 8) * 384 + n] = acc[j];
    }
    __syncthreads();
    // element-wise GRU update: 1024 (b,j) cells over 256 threads
    for (int e = threadIdx.x; e < 1024; e += blockDim.x) {
      int b = e >> 7, j = e & 127;
      const half_t* xr = xg + ((size_t)(b * L_ + l)) * 384;
      float gr = hhacc[b * 384 + j]       + bd[j];
      float gz = hhacc[b * 384 + 128 + j] + bd[128 + j];
      float gn = hhacc[b * 384 + 256 + j] + bd[256 + j];
      float r = sigmoidf_((float)xr[j] + gr);
      float z = sigmoidf_((float)xr[128 + j] + gz);
      float nn = tanhf((float)xr[256 + j] + r * gn);
      float hold = hf[e];
      int len = lens[b];
      float mt = (l < len) ? 1.f : 0.f;
      float hnew = (1.f - z) * nn + z * hold;
      float h2 = (mt > 0.f) ? hnew : hold;
      hf[e] = h2;
      h16[b * 128 + j] = (half_t)h2;
      int lo = dir ? ((l < len) ? (len - 1 - l) : l) : l;
      float ov = hnew * mt;
      size_t oidx = ((size_t)(b * L_ + lo)) * 256 + colbase + j;
      if (OUTF32) ((float*)out)[oidx] = ov;
      else        ((half_t*)out)[oidx] = (half_t)ov;
    }
    __syncthreads();
  }
}

// ---- self-attention: per (q,k) tile, scores for all 8 batches in regs, ----
// softmax over BATCH axis, query mask, per-k column sums (no score tensor).
__global__ void attn_scores_k(const half_t* __restrict__ Qh, const half_t* __restrict__ Kh,
                              const int* __restrict__ lens, float* __restrict__ colpart) {
  const int lane = threadIdx.x;
  const int hi = lane >> 4;
  const int qt = blockIdx.x, kt = blockIdx.y;
  v8f acc[8];
#pragma unroll
  for (int b = 0; b < 8; ++b) {
    const half_t* qrow = Qh + ((size_t)(b * L_ + qt * 16 + (lane & 15))) * 256;
    const half_t* krow = Kh + ((size_t)(b * L_ + kt * 16 + (lane & 15))) * 256;
    v8f a = {};
#pragma unroll
    for (int kk = 0; kk < 256; kk += 32)
      a = WMMA_F16(frag_a_f16(qrow, kk, hi), frag_b_row(krow, kk, hi), a);
    acc[b] = a;
  }
  int lenl[8];
#pragma unroll
  for (int b = 0; b < 8; ++b) lenl[b] = lens[b];
  float col[8];
#pragma unroll
  for (int b = 0; b < 8; ++b) col[b] = 0.f;
#pragma unroll
  for (int j = 0; j < 8; ++j) {
    int q = qt * 16 + j + hi * 8;
    float mx = acc[0][j];
#pragma unroll
    for (int b = 1; b < 8; ++b) mx = fmaxf(mx, acc[b][j]);
    float s = 0.f, e[8];
#pragma unroll
    for (int b = 0; b < 8; ++b) { e[b] = __expf(acc[b][j] - mx); s += e[b]; }
    float inv = 1.f / s;
#pragma unroll
    for (int b = 0; b < 8; ++b) col[b] += (q < lenl[b]) ? e[b] * inv : 0.f;
  }
#pragma unroll
  for (int b = 0; b < 8; ++b) {
    float v = col[b] + __shfl_xor(col[b], 16, 32);  // fold M=0..7 with M=8..15
    if (lane < 16)  // single-use staging buffer: keep it out of near caches
      __builtin_nontemporal_store(v, &colpart[((size_t)qt * 8 + b) * L_ + kt * 16 + lane]);
  }
}
__global__ void wsum_k(const float* __restrict__ colpart, float* __restrict__ wsum) {
  int b = blockIdx.y;
  int k = blockIdx.x * 256 + threadIdx.x;
  float s = 0.f;
  for (int qt = 0; qt < 128; ++qt)
    s += __builtin_nontemporal_load(&colpart[((size_t)qt * 8 + b) * L_ + k]);
  wsum[b * L_ + k] = s;
}
__global__ void sa_k(const float* __restrict__ wsum, const half_t* __restrict__ Vh,
                     float* __restrict__ sa) {
  int b = blockIdx.x, d = threadIdx.x;  // 256 threads
  const half_t* vb = Vh + (size_t)b * L_ * 256;
  const float* wb = wsum + b * L_;
  float s = 0.f;
  for (int k = 0; k < L_; ++k) {
    __builtin_prefetch(&vb[(size_t)(k + 8) * 256 + d], 0, 1);  // global_prefetch_b8
    s += wb[k] * (float)vb[(size_t)k * 256 + d];
  }
  sa[b * 256 + d] = s;
}

// ---- additive attention pieces -------------------------------------------
__global__ void svec_k(const float* __restrict__ Wa_w, const float* __restrict__ Wa_b,
                       const float* __restrict__ tmean, const float* __restrict__ sa,
                       float* __restrict__ svec) {
  int which = blockIdx.x, b = blockIdx.y, g = threadIdx.x;  // 256 threads
  float s = Wa_b[g];
  const float* wr = Wa_w + (size_t)g * 512;
  if (which == 0) {
    const float* tm = tmean + b * 128;
    for (int j = 0; j < 256; ++j) s += wr[j] * tm[j >> 1];  // interleaved duplicate
  } else {
    const float* sv = sa + b * 256;
    for (int j = 0; j < 256; ++j) s += wr[j] * sv[j];
  }
  svec[(which * 8 + b) * 256 + g] = s;
}
__global__ void res_k(const float* __restrict__ Wenc, const float* __restrict__ svec,
                      const float* __restrict__ VT_w, const float* __restrict__ VT_b,
                      const int* __restrict__ lens, float* __restrict__ resm) {
  int l = blockIdx.x, b = blockIdx.y, which = blockIdx.z, lane = threadIdx.x;
  const float* we = Wenc + ((size_t)(b * L_ + l)) * 256;
  const float* sv = svec + (which * 8 + b) * 256;
  float s = 0.f;
  for (int j = lane; j < 256; j += 32) s += VT_w[j] * tanhf(we[j] + sv[j]);
  for (int o = 16; o > 0; o >>= 1) s += __shfl_xor(s, o, 32);
  if (lane == 0) {
    float r = s + VT_b[0];
    resm[(which * 8 + b) * L_ + l] = (l < lens[b]) ? r : 0.f;  // score*mask BEFORE softmax
  }
}
__global__ void attn_out_k(const float* __restrict__ resm, const float* __restrict__ enc,
                           float* __restrict__ dvec) {
  int which = blockIdx.x, b = blockIdx.y, tid = threadIdx.x;  // 256 threads
  const float* r = resm + (which * 8 + b) * L_;
  __shared__ float w[L_];
  __shared__ float red[256];
  float mx = -1e30f;
  for (int l = tid; l < L_; l += 256) mx = fmaxf(mx, r[l]);
  red[tid] = mx; __syncthreads();
  for (int o = 128; o > 0; o >>= 1) { if (tid < o) red[tid] = fmaxf(red[tid], red[tid + o]); __syncthreads(); }
  mx = red[0]; __syncthreads();
  float s = 0.f;
  for (int l = tid; l < L_; l += 256) { float e = __expf(r[l] - mx); w[l] = e; s += e; }
  red[tid] = s; __syncthreads();
  for (int o = 128; o > 0; o >>= 1) { if (tid < o) red[tid] += red[tid + o]; __syncthreads(); }
  float inv = 1.f / red[0];
  __syncthreads();
  float acc = 0.f;
  const float* eb = enc + (size_t)b * L_ * 256 + tid;
  for (int l = 0; l < L_; ++l) acc += w[l] * inv * eb[(size_t)l * 256];
  dvec[(which * 8 + b) * 256 + tid] = acc;
}
__global__ void final_k(const float* __restrict__ dvec, const float* __restrict__ fc_w,
                        const float* __restrict__ fc_b, float* __restrict__ out) {
  int t = threadIdx.x;
  if (t >= 48) return;
  int b = t / 6, o = t % 6;
  const float* d0 = dvec + b * 256;
  const float* d1 = dvec + (8 + b) * 256;
  const float* w = fc_w + o * 256;
  float s = fc_b[o];
  for (int j = 0; j < 256; ++j) s += 0.5f * (d0[j] + d1[j]) * w[j];
  out[b * 6 + o] = s;
}

// ---------------------------------------------------------------------------
extern "C" void kernel_launch(void* const* d_in, const int* in_sizes, int n_in,
                              void* d_out, int out_size, void* d_ws, size_t ws_size,
                              hipStream_t stream) {
  (void)in_sizes; (void)n_in; (void)out_size; (void)ws_size;
  const int*   tok    = (const int*)d_in[0];
  const int*   lens   = (const int*)d_in[1];
  const float* embed  = (const float*)d_in[2];
  const float* wih0   = (const float*)d_in[3];
  const float* whh0   = (const float*)d_in[4];
  const float* bih0   = (const float*)d_in[5];
  const float* bhh0   = (const float*)d_in[6];
  const float* wih1   = (const float*)d_in[7];
  const float* whh1   = (const float*)d_in[8];
  const float* bih1   = (const float*)d_in[9];
  const float* bhh1   = (const float*)d_in[10];
  const float* conv_w = (const float*)d_in[11];
  const float* conv_b = (const float*)d_in[12];
  const float* Wa_w   = (const float*)d_in[13];
  const float* Wa_b   = (const float*)d_in[14];
  const float* VT_w   = (const float*)d_in[15];
  const float* VT_b   = (const float*)d_in[16];
  const float* Qw     = (const float*)d_in[17];
  const float* Kw     = (const float*)d_in[18];
  const float* Vw     = (const float*)d_in[19];
  const float* fc_w   = (const float*)d_in[20];
  const float* fc_b   = (const float*)d_in[21];
  float* out = (float*)d_out;

  char* w = (char*)d_ws;
  size_t off = 0;
  auto alloc = [&](size_t bytes) -> void* {
    void* p = w + off;
    off += (bytes + 255) & ~(size_t)255;
    return p;
  };
  half_t* cWih0 = (half_t*)alloc(2 * 384 * 128 * 2);
  half_t* cWhh0 = (half_t*)alloc(2 * 384 * 128 * 2);
  half_t* cWih1 = (half_t*)alloc(2 * 384 * 256 * 2);
  half_t* cWhh1 = (half_t*)alloc(2 * 384 * 128 * 2);
  half_t* cQw   = (half_t*)alloc(256 * 256 * 2);
  half_t* cKw   = (half_t*)alloc(256 * 256 * 2);
  half_t* cVw   = (half_t*)alloc(256 * 256 * 2);
  half_t* cWae  = (half_t*)alloc(256 * 256 * 2);
  half_t* cWc   = (half_t*)alloc(3 * 128 * 256 * 2);
  half_t* embT  = (half_t*)alloc((size_t)B_ * 128 * TS * 2);  // also conv2 output (P2)
  half_t* P1    = (half_t*)alloc((size_t)B_ * 128 * TS * 2);  // conv1 & conv3 output
  half_t* padh  = (half_t*)alloc((size_t)B_ * L_ * 128 * 2);
  half_t* xgf   = (half_t*)alloc((size_t)B_ * L_ * 384 * 2);
  half_t* xgb   = (half_t*)alloc((size_t)B_ * L_ * 384 * 2);
  half_t* h1h   = (half_t*)alloc((size_t)B_ * L_ * 256 * 2);
  float*  enc   = (float*) alloc((size_t)B_ * L_ * 256 * 4);
  half_t* Qh    = (half_t*)alloc((size_t)B_ * L_ * 256 * 2);
  half_t* Kh    = (half_t*)alloc((size_t)B_ * L_ * 256 * 2);
  half_t* Vh    = (half_t*)alloc((size_t)B_ * L_ * 256 * 2);
  float*  Wenc  = (float*) alloc((size_t)B_ * L_ * 256 * 4);
  float*  colp  = (float*) alloc((size_t)128 * 8 * L_ * 4);
  float*  wsum  = (float*) alloc((size_t)B_ * L_ * 4);
  float*  tmean = (float*) alloc(B_ * 128 * 4);
  float*  sa    = (float*) alloc(B_ * 256 * 4);
  float*  svec  = (float*) alloc(2 * B_ * 256 * 4);
  float*  resm  = (float*) alloc((size_t)2 * B_ * L_ * 4);
  float*  dvec  = (float*) alloc(2 * B_ * 256 * 4);

  auto cast = [&](const float* s, half_t* d, int n) {
    castf_k<<<(n + 255) / 256, 256, 0, stream>>>(s, d, n);
  };
  cast(wih0, cWih0, 2 * 384 * 128);
  cast(whh0, cWhh0, 2 * 384 * 128);
  cast(wih1, cWih1, 2 * 384 * 256);
  cast(whh1, cWhh1, 2 * 384 * 128);
  cast(Qw, cQw, 256 * 256);
  cast(Kw, cKw, 256 * 256);
  cast(Vw, cVw, 256 * 256);
  wae_k<<<(65536 + 255) / 256, 256, 0, stream>>>(Wa_w, cWae);
  convw_k<<<(98304 + 255) / 256, 256, 0, stream>>>(conv_w, cWc);

  // embedding
  embed_k<<<dim3(TE, B_), 128, 0, stream>>>(tok, lens, embed, embT, padh);

  // conv chain (lengths 2051 -> 2050 -> 2049 -> 2048), masked mean
  conv_wmma_k<<<dim3(129, B_), 256, 0, stream>>>(embT, cWc,         conv_b,       P1,   2050);
  conv_wmma_k<<<dim3(129, B_), 256, 0, stream>>>(P1,   cWc + 32768, conv_b + 128, embT, 2049);
  conv_wmma_k<<<dim3(128, B_), 256, 0, stream>>>(embT, cWc + 65536, conv_b + 256, P1,   2048);
  tmean_k<<<dim3(128, B_), 32, 0, stream>>>(P1, lens, tmean);

  // GRU layer 0: input-gate GEMMs + persistent scan (fwd+bwd blocks)
  gemm_wmma<0, false, true><<<dim3(1024, 24), 32, 0, stream>>>(padh, cWih0,             128, bih0,       1.f, xgf, 384, 128, lens);
  gemm_wmma<0, true,  true><<<dim3(1024, 24), 32, 0, stream>>>(padh, cWih0 + 384 * 128, 128, bih0 + 384, 1.f, xgb, 384, 128, lens);
  gru_scan<false><<<2, 256, 0, stream>>>(xgf, xgb, cWhh0, bhh0, h1h, lens);

  // GRU layer 1
  gemm_wmma<2, false, true><<<dim3(1024, 24), 32, 0, stream>>>(h1h, cWih1,             256, bih1,       1.f, xgf, 384, 256, lens);
  gemm_wmma<2, true,  true><<<dim3(1024, 24), 32, 0, stream>>>(h1h, cWih1 + 384 * 256, 256, bih1 + 384, 1.f, xgb, 384, 256, lens);
  gru_scan<true><<<2, 256, 0, stream>>>(xgf, xgb, cWhh1, bhh1, enc, lens);

  // QKV projections on inp2 = [pad, pad] (AMODE 1 duplicates features)
  const float invscale = 1.f / sqrtf(128.f);
  gemm_wmma<1, false, true><<<dim3(1024, 16), 32, 0, stream>>>(padh, cQw, 256, nullptr, invscale, Qh, 256, 256, lens);
  gemm_wmma<1, false, true><<<dim3(1024, 16), 32, 0, stream>>>(padh, cKw, 256, nullptr, 1.f,      Kh, 256, 256, lens);
  gemm_wmma<1, false, true><<<dim3(1024, 16), 32, 0, stream>>>(padh, cVw, 256, nullptr, 1.f,      Vh, 256, 256, lens);

  // streaming scores: softmax over batch, masked column sums, then sa
  attn_scores_k<<<dim3(128, 128), 32, 0, stream>>>(Qh, Kh, lens, colp);
  wsum_k<<<dim3(8, B_), 256, 0, stream>>>(colp, wsum);
  sa_k<<<B_, 256, 0, stream>>>(wsum, Vh, sa);

  // additive attention (Wa split: shared enc part via WMMA, s-part per batch)
  gemm_wmma<3, false, false><<<dim3(1024, 16), 32, 0, stream>>>(enc, cWae, 256, nullptr, 1.f, Wenc, 256, 256, lens);
  svec_k<<<dim3(2, B_), 256, 0, stream>>>(Wa_w, Wa_b, tmean, sa, svec);
  res_k<<<dim3(L_, B_, 2), 32, 0, stream>>>(Wenc, svec, VT_w, VT_b, lens, resm);
  attn_out_k<<<dim3(2, B_), 256, 0, stream>>>(resm, enc, dvec);

  final_k<<<1, 64, 0, stream>>>(dvec, fc_w, fc_b, out);
}
